// Autoencoder_20684562498273
// MI455X (gfx1250) — matrix operand order repaired and checked by reading, not compiled, not run on hardware
//
#include <hip/hip_runtime.h>
#include <hip/hip_bf16.h>

// ---------------------------------------------------------------------------
// VQ-VAE forward for MI455X (gfx1250, wave32, WMMA + Tensor Data Mover).
// All GEMM-shaped work (conv / convT implicit GEMM, VQ distance GEMM) on
// v_wmma_f32_16x16x32_bf16. Weights / codebook / zf tokens are packed to
// bf16 once per call (zf also transposed to token-major so every WMMA A
// fragment is a contiguous run); dense 2D operand tiles are DMA'd to LDS by
// the TDM (double-buffered, OOB zero-fill, per-row LDS bank padding);
// im2col gather tiles are staged by VALU with shift/mask addressing
// (spatial dims are powers of two). Fragments are b32 loads bit-cast to
// bf16 vectors.
// ---------------------------------------------------------------------------

typedef __attribute__((ext_vector_type(16))) __bf16 v16bf;
typedef __attribute__((ext_vector_type(8)))  float  v8f;
typedef unsigned int u32x4 __attribute__((ext_vector_type(4)));
typedef unsigned int u32x8 __attribute__((ext_vector_type(8)));
typedef int          i32x8 __attribute__((ext_vector_type(8)));
typedef int          i32x4 __attribute__((ext_vector_type(4)));

#define DEVFN static __device__ __forceinline__

#if defined(__has_builtin)
#  if __has_builtin(__builtin_amdgcn_tensor_load_to_lds)
#    define HAVE_TDM 1
#  else
#    define HAVE_TDM 0
#  endif
#else
#  define HAVE_TDM 0
#endif

#if HAVE_TDM
// 2D tile load Global->LDS via Tensor Data Mover (D# per ISA §8.3-8.5).
// dims/strides in elements, addresses in bytes. OOB reads return zero.
DEVFN void tdm_load_2d(unsigned lds_off, const void* gptr,
                       unsigned tensor_d0, unsigned tensor_d1,
                       unsigned tile_d0, unsigned tile_d1,
                       unsigned long long stride0,
                       unsigned data_size_code,          // 0=1B,1=2B,2=4B
                       bool pad_en, unsigned pad_interval_code,
                       unsigned pad_amount_code)
{
  unsigned long long ga = (unsigned long long)gptr;
  u32x4 g0;
  g0[0] = 1u;                                              // count=1 (user)
  g0[1] = lds_off;                                         // lds_addr
  g0[2] = (unsigned)(ga & 0xffffffffull);                  // global_addr lo
  g0[3] = (unsigned)((ga >> 32) & 0x01ffffffull) | (2u << 30); // hi | type=2
  unsigned w0 = (data_size_code << 16);
  if (pad_en) w0 |= (1u << 20) | (pad_interval_code << 22) | (pad_amount_code << 25);
  i32x8 g1;
  g1[0] = (int)w0;                                         // mask=0, flags
  g1[1] = (int)((tensor_d0 & 0xffffu) << 16);              // abar=0 | td0 lo
  g1[2] = (int)((tensor_d0 >> 16) | ((tensor_d1 & 0xffffu) << 16));
  g1[3] = (int)((tensor_d1 >> 16) | ((tile_d0 & 0xffffu) << 16));
  g1[4] = (int)(tile_d1 & 0xffffu);                        // tile_dim2=0
  g1[5] = (int)(unsigned)(stride0 & 0xffffffffull);        // dim0 stride lo
  g1[6] = (int)(unsigned)((stride0 >> 32) & 0xffffull);    // hi | s1 lo(=0)
  g1[7] = 0;
  #pragma unroll
  for (int i = 0; i < 4; ++i) g0[i] = (unsigned)__builtin_amdgcn_readfirstlane((int)g0[i]);
  #pragma unroll
  for (int i = 0; i < 8; ++i) g1[i] = __builtin_amdgcn_readfirstlane(g1[i]);
  i32x4 z4 = {0, 0, 0, 0};
#if __has_include(<hip/amd_detail/amd_gfx1250_TDM.h>)
  i32x8 z8 = {0, 0, 0, 0, 0, 0, 0, 0};
  __builtin_amdgcn_tensor_load_to_lds(g0, g1, z4, z4, z8, 0);
#else
  __builtin_amdgcn_tensor_load_to_lds(g0, g1, z4, z4, 0);
#endif
}
#endif // HAVE_TDM

DEVFN unsigned lds_addr32(const void* p) {
  return (unsigned)(unsigned long long)p;   // LDS aperture: addr[31:0] = offset
}

// 16 consecutive bf16 (32B, 4B-aligned) -> fragment, via b32 loads.
DEVFN v16bf frag16(const __bf16* p) {
  u32x8 a;
  const unsigned* q = (const unsigned*)p;
  #pragma unroll
  for (int j = 0; j < 8; ++j) a[j] = q[j];
  return __builtin_bit_cast(v16bf, a);
}
// two 8-bf16 runs (A fragment halves: K lo / K hi groups)
DEVFN v16bf frag8x2(const __bf16* p0, const __bf16* p1) {
  u32x8 a;
  const unsigned* q0 = (const unsigned*)p0;
  const unsigned* q1 = (const unsigned*)p1;
  #pragma unroll
  for (int j = 0; j < 4; ++j) { a[j] = q0[j]; a[4 + j] = q1[j]; }
  return __builtin_bit_cast(v16bf, a);
}

// ================= unified implicit-GEMM conv / convT (stride 2) ===========
// GEMM: M=Cout, N=B*Ho*Wo, K=Cin*9. WG=128 thr=4 waves along M.
// WG tile 64M x 32N; per wave 16M x 32N (two accumulators).
// A slab (bf16, pre-packed [Cout][Cin*9]) streams via double-buffered TDM.
// TRANS=false: x[ih=2oh-1+kh]; TRANS=true: parity-gated transposed conv.
template <bool TRANS>
__global__ __launch_bounds__(128) void conv_gemm_wmma(
    const float* __restrict__ x, const __bf16* __restrict__ wbf,
    const float* __restrict__ bias, float* __restrict__ y,
    int Cin, int Cout, int Hin, int Win, int Hout, int Wout,
    int whs /*log2(Hout*Wout)*/, int wls /*log2(Wout)*/, int Bn)
{
  const int Ktot = Cin * 9;
  const int HW   = 1 << whs;
  const int Ntot = Bn << whs;

  const int nbase = blockIdx.x * 32;
  const int mbase = blockIdx.y * 64;
  const int tid  = threadIdx.x;
  const int wave = tid >> 5;
  const int lane = tid & 31;
  const int mrow0 = mbase + wave * 16;

  __shared__ __bf16 ldsA[2][64 * 34];  // TDM row pad: 1 DWORD per 16 DWORDs
  __shared__ __bf16 ldsB[32][34];

  v8f acc0 = {}, acc1 = {};
  const int kfix  = tid & 31;          // (tid+128*i)&31 is constant
  const int cbase = tid >> 5;
  const int nsteps = (Ktot + 31) >> 5;

#if HAVE_TDM
  if (wave == 0) {
    tdm_load_2d(lds_addr32(&ldsA[0][0]), wbf + (size_t)mbase * Ktot,
                (unsigned)Ktot, (unsigned)(Cout - mbase),
                32u, 64u, (unsigned long long)Ktot,
                1u /*2B*/, true, 3u /*16 DWORDs*/, 0u /*1 DWORD pad*/);
  }
#endif

  for (int s = 0; s < nsteps; ++s) {
    const int kb = s << 5;
#if !HAVE_TDM
    for (int e = tid; e < 64 * 32; e += 128) {
      int row = e >> 5, k = e & 31;
      __bf16 v = (__bf16)0.f;
      if (mbase + row < Cout && kb + k < Ktot)
        v = wbf[(size_t)(mbase + row) * Ktot + kb + k];
      ldsA[s & 1][row * 34 + k] = v;
    }
#endif
    // ---- cooperative im2col of the 32K x 32N B tile (overlaps the DMA) ----
    {
      int kk = kb + kfix;
      bool kok = kk < Ktot;
      int ci = kk / 9, rem = kk - ci * 9;      // one /9 per K-step per thread
      int kh = rem / 3, kw = rem - kh * 3;
      #pragma unroll
      for (int i = 0; i < 8; ++i) {
        int c = cbase + 4 * i;
        int col = nbase + c;
        float v = 0.f;
        if (kok && col < Ntot) {
          int n  = col >> whs, r2 = col & (HW - 1);
          int oh = r2 >> wls,  ow = r2 & (Wout - 1);
          if (TRANS) {
            int ih2 = oh + 1 - kh;
            int iw2 = ow + 1 - kw;
            if (((ih2 | iw2) >= 0) && !((ih2 | iw2) & 1)) {
              int ih = ih2 >> 1, iw = iw2 >> 1;
              if (ih < Hin && iw < Win)
                v = x[((n * Cin + ci) * Hin + ih) * Win + iw];
            }
          } else {
            int ih = oh * 2 - 1 + kh;
            int iw = ow * 2 - 1 + kw;
            if ((unsigned)ih < (unsigned)Hin && (unsigned)iw < (unsigned)Win)
              v = x[((n * Cin + ci) * Hin + ih) * Win + iw];
          }
        }
        ldsB[c][kfix] = (__bf16)v;
      }
    }
#if HAVE_TDM
    if (wave == 0) __builtin_amdgcn_s_wait_tensorcnt((unsigned short)0);
#endif
    __syncthreads();
#if HAVE_TDM
    // pipeline: DMA next K-slab into the other buffer while we compute
    if (wave == 0 && s + 1 < nsteps) {
      int kn = kb + 32;
      tdm_load_2d(lds_addr32(&ldsA[(s + 1) & 1][0]),
                  wbf + (size_t)mbase * Ktot + kn,
                  (unsigned)(Ktot - kn), (unsigned)(Cout - mbase),
                  32u, 64u, (unsigned long long)Ktot,
                  1u, true, 3u, 0u);
    }
#endif
    // A fragment (contiguous K order, matches B): lane<16 K{0..15},
    // lane>=16 K{16..31}
    const __bf16* ar = &ldsA[s & 1][(wave * 16 + (lane & 15)) * 34];
    int koffA = (lane < 16) ? 0 : 16;
    v16bf afrag = frag16(ar + koffA);
    // two B fragments (cols 0..15 / 16..31): contiguous 16-bf16 runs
    int cc    = lane & 15;
    int koffB = (lane < 16) ? 0 : 16;
    v16bf bf0 = frag16(&ldsB[cc][koffB]);
    v16bf bf1 = frag16(&ldsB[16 + cc][koffB]);

    acc0 = __builtin_amdgcn_wmma_f32_16x16x32_bf16(false, afrag, false, bf0,
                                                   (short)0, acc0, false, false);
    acc1 = __builtin_amdgcn_wmma_f32_16x16x32_bf16(false, afrag, false, bf1,
                                                   (short)0, acc1, false, false);
    __syncthreads();
  }

  // store: C/D layout lane<16 rows 0..7, lane>=16 rows 8..15
  #pragma unroll
  for (int half = 0; half < 2; ++half) {
    int col = nbase + half * 16 + (lane & 15);
    if (col < Ntot) {
      int n  = col >> whs, r2 = col & (HW - 1);
      int oh = r2 >> wls,  ow = r2 & (Wout - 1);
      const v8f& a = half ? acc1 : acc0;
      #pragma unroll
      for (int r = 0; r < 8; ++r) {
        int mrow = mrow0 + (lane >> 4) * 8 + r;
        if (mrow < Cout)
          y[((n * Cout + mrow) * Hout + oh) * Wout + ow] = a[r] + bias[mrow];
      }
    }
  }
}

// ======================= weight / data packing to bf16 =====================
__global__ __launch_bounds__(256) void cvt_bf16_kernel(
    const float* __restrict__ in, __bf16* __restrict__ out, int n)
{
  int i = blockIdx.x * 256 + threadIdx.x;
  if (i < n) out[i] = (__bf16)in[i];
}

// dec weights (Cin,Cout,3,3) -> A layout [Cout][Cin*9] bf16
__global__ __launch_bounds__(256) void pack_w_dec_kernel(
    const float* __restrict__ w, __bf16* __restrict__ out, int Cin, int Cout)
{
  int Ktot = Cin * 9;
  int i = blockIdx.x * 256 + threadIdx.x;
  if (i < Cout * Ktot) {
    int co = i / Ktot, kk = i - co * Ktot;
    int ci = kk / 9,   r  = kk - ci * 9;
    out[i] = (__bf16)w[(ci * Cout + co) * 9 + r];
  }
}

// encoder output (16,512,16,16) NCHW -> token-major zf [4096][512] bf16,
// so VQ A fragments are contiguous K runs.
__global__ __launch_bounds__(256) void pack_zf_kernel(
    const float* __restrict__ h, __bf16* __restrict__ out)
{
  int i = blockIdx.x * 256 + threadIdx.x;    // over 4096*512
  if (i < 4096 * 512) {
    int t = i >> 9, c = i & 511;
    int n = t >> 8, oh = (t >> 4) & 15, ow = t & 15;
    out[i] = (__bf16)h[((n * 512 + c) * 16 + oh) * 16 + ow];
  }
}

// ======================= BatchNorm statistics ==============================
__global__ __launch_bounds__(256) void bn_stats(
    const float* __restrict__ y, float* __restrict__ mean,
    float* __restrict__ rvar, int C, int whs, int Bn)
{
  int c = blockIdx.x;
  const int HW = 1 << whs;
  const int total = Bn << whs;
  float s = 0.f, s2 = 0.f;
  for (int i = threadIdx.x; i < total; i += 256) {
    int n = i >> whs, r = i & (HW - 1);
    float v = y[((size_t)(n * C + c) << whs) + r];
    s += v; s2 += v * v;
  }
  __shared__ float rs[256], rq[256];
  rs[threadIdx.x] = s; rq[threadIdx.x] = s2;
  __syncthreads();
  for (int off = 128; off > 0; off >>= 1) {
    if (threadIdx.x < off) {
      rs[threadIdx.x] += rs[threadIdx.x + off];
      rq[threadIdx.x] += rq[threadIdx.x + off];
    }
    __syncthreads();
  }
  if (threadIdx.x == 0) {
    float inv = 1.f / (float)total;
    float m = rs[0] * inv;
    float var = rq[0] * inv - m * m;
    mean[c] = m;
    rvar[c] = rsqrtf(var + 1e-5f);
  }
}

// ======================= fused BN + exact GELU =============================
__global__ __launch_bounds__(256) void bn_gelu(
    float* __restrict__ y, const float* __restrict__ mean,
    const float* __restrict__ rvar, const float* __restrict__ g,
    const float* __restrict__ be, int C, int whs, int total)
{
  int i = blockIdx.x * 256 + threadIdx.x;
  if (i < total) {
    int c = (i >> whs) % C;
    float xh = (y[i] - mean[c]) * rvar[c] * g[c] + be[c];
    y[i] = 0.5f * xh * (1.f + erff(xh * 0.70710678118654752f));
  }
}

// ======================= VQ: codebook norms ================================
__global__ __launch_bounds__(256) void vq_cnorm(
    const float* __restrict__ cb, float* __restrict__ cnorm)
{
  int e = blockIdx.x * 256 + threadIdx.x;
  if (e < 1024) {
    const float* r = cb + (size_t)e * 512;
    float s = 0.f;
    for (int c = 0; c < 512; ++c) { float v = r[c]; s += v * v; }
    cnorm[e] = s;
  }
}

// ======================= VQ: WMMA score GEMM + argmin ======================
// 4096 tokens x 1024 codes x 512 dims, all bf16 operands. WG owns 16 tokens:
// token-major zf tile DMA'd by one TDM op with per-row bank padding (row
// stride 514 bf16 = 257 banks, conflict-free); waves interleave over the 64
// code tiles; codebook rows stream from L2 as 32B contiguous runs.
__global__ __launch_bounds__(128) void vq_scores_wmma(
    const __bf16* __restrict__ zf2, const __bf16* __restrict__ cbh,
    const float* __restrict__ cnorm, int* __restrict__ bestidx)
{
  const int tbase = blockIdx.x * 16;
  const int tid = threadIdx.x, wave = tid >> 5, lane = tid & 31;

  __shared__ __bf16 ldsZ[16 * 514];   // [token][512 + 2 pad]

#if HAVE_TDM
  if (wave == 0) {
    tdm_load_2d(lds_addr32(ldsZ), zf2 + (size_t)tbase * 512,
                512u, 16u, 512u, 16u, 512ull,
                1u /*2B*/, true, 7u /*256 DWORDs*/, 0u /*1 DWORD pad*/);
    __builtin_amdgcn_s_wait_tensorcnt((unsigned short)0);
  }
#else
  for (int e = tid; e < 16 * 512; e += 128) {
    int t = e >> 9, c = e & 511;
    ldsZ[t * 514 + c] = zf2[(size_t)(tbase + t) * 512 + c];
  }
#endif
  __syncthreads();

  float bestv[8]; int besti[8];
  #pragma unroll
  for (int r = 0; r < 8; ++r) { bestv[r] = 3.4e38f; besti[r] = 0; }

  const int tt = lane & 15;
  const __bf16* zr = &ldsZ[tt * 514];
  const int koffA = (lane < 16) ? 0 : 8;
  const int koffB = (lane < 16) ? 0 : 16;

  for (int j = 0; j < 16; ++j) {
    int ebase = (wave + 4 * j) * 16;
    if (j + 1 < 16)
      __builtin_prefetch(cbh + (size_t)((wave + 4 * (j + 1)) * 16) * 512, 0, 1);
    v8f acc = {};
    for (int kb = 0; kb < 512; kb += 32) {
      v16bf afrag = frag16(zr + kb + koffB);
      v16bf bfrag = frag16(cbh + (size_t)(ebase + tt) * 512 + kb + koffB);
      acc = __builtin_amdgcn_wmma_f32_16x16x32_bf16(false, afrag, false, bfrag,
                                                    (short)0, acc, false, false);
    }
    int e = ebase + tt;
    float cn = cnorm[e];
    #pragma unroll
    for (int r = 0; r < 8; ++r) {
      float d = cn - 2.f * acc[r];     // |z|^2 per-token constant: argmin-invariant
      if (d < bestv[r]) { bestv[r] = d; besti[r] = e; }
    }
  }

  __shared__ float rv[16][64];
  __shared__ int   ri[16][64];
  int slot  = wave * 16 + tt;
  int rbase = (lane >> 4) * 8;
  #pragma unroll
  for (int r = 0; r < 8; ++r) { rv[rbase + r][slot] = bestv[r]; ri[rbase + r][slot] = besti[r]; }
  __syncthreads();
  if (tid < 16) {
    float bv = 3.4e38f; int bi = 0;
    for (int s = 0; s < 64; ++s) {
      float v = rv[tid][s];
      if (v < bv) { bv = v; bi = ri[tid][s]; }
    }
    bestidx[tbase + tid] = bi;
  }
}

// ======================= VQ: gather + loss =================================
// Forward value of zq_st is exactly zq; loss = 1.25 * mean((zf - zq)^2).
// Uses fp32 zf and fp32 codebook for loss fidelity.
__global__ __launch_bounds__(256) void vq_gather_loss(
    const float* __restrict__ h, const float* __restrict__ cb,
    const int* __restrict__ bestidx, float* __restrict__ zq,
    float* __restrict__ loss)
{
  int t = blockIdx.x;
  int n = t >> 8, oh = (t >> 4) & 15, ow = t & 15;
  int e = bestidx[t];
  const float* cr = cb + (size_t)e * 512;
  float s = 0.f;
  for (int c = threadIdx.x; c < 512; c += 256) {
    int idx = ((n * 512 + c) * 16 + oh) * 16 + ow;
    float zf = h[idx];
    float q  = cr[c];
    float d  = zf - q;
    s += d * d;
    zq[idx] = q;
  }
  __shared__ float red[256];
  red[threadIdx.x] = s;
  __syncthreads();
  for (int off = 128; off > 0; off >>= 1) {
    if (threadIdx.x < off) red[threadIdx.x] += red[threadIdx.x + off];
    __syncthreads();
  }
  if (threadIdx.x == 0)
    atomicAdd(loss, red[0] * (1.25f / (4096.f * 512.f)));
}

// ======================= misc elementwise ==================================
__global__ void zero_scalar(float* p) { if (threadIdx.x == 0 && blockIdx.x == 0) *p = 0.f; }

__global__ __launch_bounds__(256) void tanh_out(
    const float* __restrict__ in, float* __restrict__ out, int n,
    const float* __restrict__ loss)
{
  int i = blockIdx.x * 256 + threadIdx.x;
  if (i < n) out[i] = tanhf(in[i]);
  if (i == 0) out[n] = *loss;
}

// ======================= host-side orchestration ===========================
extern "C" void kernel_launch(void* const* d_in, const int* in_sizes, int n_in,
                              void* d_out, int out_size, void* d_ws, size_t ws_size,
                              hipStream_t stream)
{
  (void)in_sizes; (void)n_in; (void)out_size; (void)ws_size;

  const float* x = (const float*)d_in[0];
  const float *ew[4], *eb[4], *eg[4], *ebe[4];
  for (int i = 0; i < 4; ++i) {
    ew[i]  = (const float*)d_in[1 + 4 * i];
    eb[i]  = (const float*)d_in[2 + 4 * i];
    eg[i]  = (const float*)d_in[3 + 4 * i];
    ebe[i] = (const float*)d_in[4 + 4 * i];
  }
  const float* cb = (const float*)d_in[17];
  const float *dw[4], *db[4], *dg[4], *dbe[4];
  for (int i = 0; i < 4; ++i) {
    dw[i]  = (const float*)d_in[18 + 4 * i];
    db[i]  = (const float*)d_in[19 + 4 * i];
    dg[i]  = (const float*)d_in[20 + 4 * i];
    dbe[i] = (const float*)d_in[21 + 4 * i];
  }

  float* ws = (float*)d_ws;
  const long BUF = 16L * 64 * 128 * 128;     // largest activation (floats)
  float* bufA  = ws;
  float* bufB  = ws + BUF;
  float* meanp = ws + 2 * BUF;
  float* rvarp = meanp + 512;
  float* cnrm  = rvarp + 512;
  int*   bidx  = (int*)(cnrm + 1024);
  float* lossp = (float*)(bidx + 4096);

  // bf16 region (packed operands)
  const int Bn = 16;
  const int ECH[5] = {3, 64, 128, 256, 512};
  const int DCH[5] = {512, 256, 128, 64, 3};
  __bf16* bfbase = (__bf16*)(lossp + 4);
  __bf16 *wench[4], *wdech[4];
  {
    __bf16* p = bfbase;
    for (int i = 0; i < 4; ++i) { wench[i] = p; p += (long)ECH[i + 1] * ECH[i] * 9; }
    for (int i = 0; i < 4; ++i) { wdech[i] = p; p += (long)DCH[i + 1] * DCH[i] * 9; }
  }
  __bf16* cbh = wdech[3] + (long)DCH[4] * DCH[3] * 9;
  __bf16* zf2 = cbh + 1024L * 512;           // token-major [4096][512]

  auto cdiv = [](long a, long b) { return (unsigned)((a + b - 1) / b); };
  auto lg2  = [](int v) { int s = 0; while ((1 << s) < v) ++s; return s; };

  // ---- pack weights + codebook to bf16 (tiny; L2-resident afterwards) ----
  for (int i = 0; i < 4; ++i) {
    int n = ECH[i + 1] * ECH[i] * 9;
    cvt_bf16_kernel<<<cdiv(n, 256), 256, 0, stream>>>(ew[i], wench[i], n);
  }
  for (int i = 0; i < 4; ++i) {
    int Cin = DCH[i], Cout = DCH[i + 1];
    pack_w_dec_kernel<<<cdiv((long)Cout * Cin * 9, 256), 256, 0, stream>>>(
        dw[i], wdech[i], Cin, Cout);
  }
  cvt_bf16_kernel<<<cdiv(1024L * 512, 256), 256, 0, stream>>>(cb, cbh, 1024 * 512);

  float* bufs[2] = {bufA, bufB};

  // ---------------- encoder ----------------
  const float* cur = x;
  for (int i = 0; i < 4; ++i) {
    int Cin = ECH[i], Cout = ECH[i + 1];
    int Hin = 256 >> i, Hout = 128 >> i;
    int wls = lg2(Hout), whs = 2 * wls;
    float* out = bufs[i & 1];
    long Ntot = (long)Bn * Hout * Hout;
    dim3 grid(cdiv(Ntot, 32), cdiv(Cout, 64));
    conv_gemm_wmma<false><<<grid, 128, 0, stream>>>(cur, wench[i], eb[i], out,
                                                    Cin, Cout, Hin, Hin, Hout, Hout,
                                                    whs, wls, Bn);
    bn_stats<<<Cout, 256, 0, stream>>>(out, meanp, rvarp, Cout, whs, Bn);
    int total = (int)((long)Bn * Cout * Hout * Hout);
    bn_gelu<<<cdiv(total, 256), 256, 0, stream>>>(out, meanp, rvarp, eg[i], ebe[i],
                                                  Cout, whs, total);
    cur = out;
  }
  // encoder output in bufB (16,512,16,16)

  // ---------------- vector quantization ----------------
  zero_scalar<<<1, 64, 0, stream>>>(lossp);
  vq_cnorm<<<4, 256, 0, stream>>>(cb, cnrm);
  pack_zf_kernel<<<cdiv(4096L * 512, 256), 256, 0, stream>>>(bufB, zf2);
  vq_scores_wmma<<<256, 128, 0, stream>>>(zf2, cbh, cnrm, bidx);
  vq_gather_loss<<<4096, 256, 0, stream>>>(bufB, cb, bidx, bufA, lossp);
  // quantized NCHW tensor in bufA

  // ---------------- decoder ----------------
  cur = bufA;
  for (int i = 0; i < 4; ++i) {
    int Cin = DCH[i], Cout = DCH[i + 1];
    int Hin = 16 << i, Hout = 32 << i;
    int wls = lg2(Hout), whs = 2 * wls;
    float* out = bufs[(i + 1) & 1];
    long Ntot = (long)Bn * Hout * Hout;
    dim3 grid(cdiv(Ntot, 32), cdiv(Cout, 64));
    conv_gemm_wmma<true><<<grid, 128, 0, stream>>>(cur, wdech[i], db[i], out,
                                                   Cin, Cout, Hin, Hin, Hout, Hout,
                                                   whs, wls, Bn);
    bn_stats<<<Cout, 256, 0, stream>>>(out, meanp, rvarp, Cout, whs, Bn);
    int total = (int)((long)Bn * Cout * Hout * Hout);
    bn_gelu<<<cdiv(total, 256), 256, 0, stream>>>(out, meanp, rvarp, dg[i], dbe[i],
                                                  Cout, whs, total);
    cur = out;
  }
  // final activations in bufA (16,3,256,256)

  const int NOUT = 16 * 3 * 256 * 256;
  tanh_out<<<cdiv(NOUT + 1, 256), 256, 0, stream>>>(bufA, (float*)d_out, NOUT, lossp);
}
